// NovelHybridRNAModel_33174327394475
// MI455X (gfx1250) — compile-verified, matrix operand order checked by reading, not run on hardware
//
#include <hip/hip_runtime.h>
#include <hip/hip_fp16.h>

typedef _Float16 f16;
typedef __attribute__((ext_vector_type(16))) _Float16 v16h;
typedef __attribute__((ext_vector_type(8)))  float    v8f;

#define DEV __device__ __forceinline__

DEV float gelu_f(float x) { return 0.5f * x * (1.0f + erff(x * 0.7071067811865476f)); }

union HV { int4 i[2]; v16h h; };

// Load one 16-lane WMMA fragment slice: 8 halves at kbase, 8 halves at kbase+16
// (per CDNA5 16-bit A/B VGPR layout; kbase is a multiple of 8 halves -> 16B aligned)
DEV v16h load_frag(const int4* rowPtr, int kbase) {
  HV u;
  u.i[0] = rowPtr[kbase >> 3];
  u.i[1] = rowPtr[(kbase >> 3) + 2];
  return u.h;
}

// ---------------------------------------------------------------------------
// Generic batched WMMA GEMM: C[m,n] = scale * sum_k A[m,k]*B[n,k]  (+ epilogue)
// NT (tiles of 16 columns per wave) is a template parameter so the k-loop is
// branch-free.  All NT B-fragments are loaded before the WMMA burst so the
// loads clause together and waits are partial instead of 0-before-each-WMMA.
// ---------------------------------------------------------------------------
enum { F_OUTF16 = 1, F_TRANS = 2, F_ACCUM = 4, F_RESID = 8,
       F_SCATTER = 16, F_GELU = 32, F_QPAD = 64 };

struct GemmP {
  const f16* A; const f16* B;
  float* Cf; f16* Ch;
  const float* bias; const float* resid; const int* sidx; const float* coord;
  int M, N, K, lda, ldb, ldc;
  long long sAin, sAout, sBin, sBout, sCin, sCout;
  int zinner;
  float scale; int flags;
};

template<int NT>
__global__ __launch_bounds__(256) void gemm16(GemmP p) {
  int z = blockIdx.z;
  int zi = z % p.zinner, zo = z / p.zinner;
  const f16* A = p.A + (size_t)zi * p.sAin + (size_t)zo * p.sAout;
  const f16* B = p.B + (size_t)zi * p.sBin + (size_t)zo * p.sBout;
  long long coff = (long long)zi * p.sCin + (long long)zo * p.sCout;

  int wave = threadIdx.x >> 5, lane = threadIdx.x & 31;
  int half = lane >> 4, l = lane & 15;
  int m0 = blockIdx.x * 128 + wave * 16;
  int n0 = blockIdx.y * (NT * 16);

  const int4* Arow = (const int4*)(A + (size_t)(m0 + l) * p.lda);
  const int4* Brow[NT];
#pragma unroll
  for (int t = 0; t < NT; t++)
    Brow[t] = (const int4*)(B + (size_t)(n0 + t * 16 + l) * p.ldb);

  v8f acc[NT] = {};
  int koff = half * 8;
  int K = p.K;
  for (int k = 0; k < K; k += 32) {
    v16h a = load_frag(Arow, k + koff);
    // speculative prefetch of next A k-block (silently dropped if OOB)
    __builtin_prefetch((const void*)(Arow + ((k + 32 + koff) >> 3)), 0, 1);
    v16h bf[NT];
#pragma unroll
    for (int t = 0; t < NT; t++) bf[t] = load_frag(Brow[t], k + koff);
#pragma unroll
    for (int t = 0; t < NT; t++)
      acc[t] = __builtin_amdgcn_wmma_f32_16x16x32_f16(
          false, a, false, bf[t], (short)0, acc[t], false, false);
  }

#pragma unroll
  for (int t = 0; t < NT; t++) {
    int col = n0 + t * 16 + l;
    float bc = p.bias ? p.bias[col] : 0.f;
#pragma unroll
    for (int j = 0; j < 8; j++) {
      int row = m0 + j + half * 8;
      float v = acc[t][j] * p.scale + bc;
      if (p.flags & F_QPAD) v += p.coord[row * 3 + (col % 3)];
      if (p.flags & F_GELU) v = gelu_f(v);
      if (p.flags & F_RESID) v += p.resid[(size_t)row * p.ldc + col];
      if (p.flags & F_SCATTER) {
        atomicAdd(&p.Cf[(size_t)p.sidx[row] * p.ldc + col], v);
      } else if (p.flags & F_OUTF16) {
        int oc = (p.flags & F_QPAD) ? ((col / 48) * 64 + col % 48) : col;
        size_t idx = (p.flags & F_TRANS) ? ((size_t)oc * p.ldc + row)
                                         : ((size_t)row * p.ldc + oc);
        p.Ch[coff + idx] = (f16)v;
      } else {
        size_t idx = (size_t)row * p.ldc + col;
        if (p.flags & F_ACCUM) v += p.Cf[coff + idx];
        p.Cf[coff + idx] = v;
      }
    }
  }
}

// ---------------------------------------------------------------------------
// Fused GEMM (+bias) + row-LayerNorm + GELU + f16 store.  N fixed to 1024:
// one block owns one 16-row stripe across all 1024 columns (8 waves x 8 tiles)
// so LN statistics reduce in LDS.
// ---------------------------------------------------------------------------
__global__ __launch_bounds__(256) void gemm_rowln(
    const f16* __restrict__ A, const f16* __restrict__ Bw,
    const float* __restrict__ bias, const float* __restrict__ lng,
    const float* __restrict__ lnb, f16* __restrict__ Out, int K, int lda) {
  __shared__ float s_sum[16], s_sq[16];
  int wave = threadIdx.x >> 5, lane = threadIdx.x & 31;
  int half = lane >> 4, l = lane & 15;
  int m0 = blockIdx.x * 16;
  if (threadIdx.x < 16) { s_sum[threadIdx.x] = 0.f; s_sq[threadIdx.x] = 0.f; }
  __syncthreads();

  const int4* Arow = (const int4*)(A + (size_t)(m0 + l) * lda);
  const int4* Brow[8];
#pragma unroll
  for (int t = 0; t < 8; t++)
    Brow[t] = (const int4*)(Bw + (size_t)(wave * 128 + t * 16 + l) * lda);

  v8f acc[8] = {};
  int koff = half * 8;
  for (int k = 0; k < K; k += 32) {
    v16h a = load_frag(Arow, k + koff);
    __builtin_prefetch((const void*)(Arow + ((k + 32 + koff) >> 3)), 0, 1);
    v16h bf[8];
#pragma unroll
    for (int t = 0; t < 8; t++) bf[t] = load_frag(Brow[t], k + koff);
#pragma unroll
    for (int t = 0; t < 8; t++)
      acc[t] = __builtin_amdgcn_wmma_f32_16x16x32_f16(
          false, a, false, bf[t], (short)0, acc[t], false, false);
  }

  float vals[8][8];
  float psum[8] = {}, psq[8] = {};
#pragma unroll
  for (int t = 0; t < 8; t++) {
    int col = wave * 128 + t * 16 + l;
    float bc = bias[col];
#pragma unroll
    for (int j = 0; j < 8; j++) {
      float v = acc[t][j] + bc;
      vals[t][j] = v; psum[j] += v; psq[j] += v * v;
    }
  }
#pragma unroll
  for (int j = 0; j < 8; j++) {
    int row = j + half * 8;
    atomicAdd(&s_sum[row], psum[j]);
    atomicAdd(&s_sq[row], psq[j]);
  }
  __syncthreads();
#pragma unroll
  for (int j = 0; j < 8; j++) {
    int row = j + half * 8;
    float mu = s_sum[row] * (1.0f / 1024.0f);
    float var = s_sq[row] * (1.0f / 1024.0f) - mu * mu;
    float rstd = rsqrtf(var + 1e-5f);
#pragma unroll
    for (int t = 0; t < 8; t++) {
      int col = wave * 128 + t * 16 + l;
      float v = (vals[t][j] - mu) * rstd * lng[col] + lnb[col];
      Out[(size_t)(m0 + row) * 1024 + col] = (f16)gelu_f(v);
    }
  }
}

// ---------------------------------------------------------------------------
// Small elementwise / reduction kernels
// ---------------------------------------------------------------------------
__global__ void cvt_w(const float* __restrict__ W, f16* __restrict__ Wt,
                      int K, int Nc, int Kpad) {
  int i = blockIdx.x * blockDim.x + threadIdx.x;
  if (i >= Nc * Kpad) return;
  int n = i / Kpad, k = i % Kpad;
  Wt[i] = (f16)((k < K) ? W[(size_t)k * Nc + n] : 0.f);
}

__global__ void cvt_f32_f16(const float* __restrict__ in, f16* __restrict__ out, int n) {
  int i = blockIdx.x * blockDim.x + threadIdx.x;
  if (i < n) out[i] = (f16)in[i];
}

__global__ void build_mi(const float* __restrict__ x, const float* __restrict__ coords,
                         const int* __restrict__ eidx, const int* __restrict__ etyp,
                         const float* __restrict__ eemb, f16* __restrict__ mi) {
  int e = blockIdx.x;
  int src = eidx[e];           // edge_index[0] = row (source)
  int dst = eidx[65536 + e];   // edge_index[1] = col (target)
  float dx = coords[src * 3 + 0] - coords[dst * 3 + 0];
  float dy = coords[src * 3 + 1] - coords[dst * 3 + 1];
  float dz = coords[src * 3 + 2] - coords[dst * 3 + 2];
  float ed = sqrtf(dx * dx + dy * dy + dz * dz);
  int et = etyp[e];
  f16* row = mi + (size_t)e * 1088;
  for (int c = threadIdx.x; c < 1088; c += blockDim.x) {
    float v;
    if (c < 512)       v = x[(size_t)dst * 512 + c];          // x_i = x[col]
    else if (c < 1024) v = x[(size_t)src * 512 + (c - 512)];  // x_j = x[row]
    else if (c < 1056) v = eemb[et * 32 + (c - 1024)];
    else if (c == 1056) v = ed;
    else v = 0.f;                                             // K pad
    row[c] = (f16)v;
  }
}

__global__ void build_xcat(const float* __restrict__ x, const float* __restrict__ agg,
                           f16* __restrict__ out) {
  int i = blockIdx.x * blockDim.x + threadIdx.x;
  if (i >= 2048 * 1024) return;
  int r = i >> 10, c = i & 1023;
  float v = (c < 512) ? x[r * 512 + c] : agg[r * 512 + (c - 512)];
  out[i] = (f16)v;
}

__global__ __launch_bounds__(256) void ln_rows(
    const float* __restrict__ X, const float* __restrict__ g,
    const float* __restrict__ b, float* outF, f16* outH, int W) {
  __shared__ float s1[256], s2[256];
  int row = blockIdx.x;
  const float* xr = X + (size_t)row * W;
  float ls = 0.f, lq = 0.f;
  for (int c = threadIdx.x; c < W; c += 256) { float v = xr[c]; ls += v; lq += v * v; }
  s1[threadIdx.x] = ls; s2[threadIdx.x] = lq; __syncthreads();
  for (int o = 128; o > 0; o >>= 1) {
    if (threadIdx.x < o) { s1[threadIdx.x] += s1[threadIdx.x + o]; s2[threadIdx.x] += s2[threadIdx.x + o]; }
    __syncthreads();
  }
  float mu = s1[0] / W, var = s2[0] / W - mu * mu, rstd = rsqrtf(var + 1e-5f);
  for (int c = threadIdx.x; c < W; c += 256) {
    float v = (xr[c] - mu) * rstd * g[c] + b[c];
    if (outF) outF[(size_t)row * W + c] = v;
    else      outH[(size_t)row * W + c] = (f16)v;
  }
}

__global__ __launch_bounds__(256) void softmax_ca(const float* __restrict__ S,
                                                  f16* __restrict__ A) {
  __shared__ float red[256];
  size_t row = blockIdx.x;
  const float* sr = S + row * 2048;
  float v[8]; float mx = -3.4e38f;
#pragma unroll
  for (int i = 0; i < 8; i++) { v[i] = sr[threadIdx.x + i * 256]; mx = fmaxf(mx, v[i]); }
  red[threadIdx.x] = mx; __syncthreads();
  for (int o = 128; o > 0; o >>= 1) {
    if (threadIdx.x < o) red[threadIdx.x] = fmaxf(red[threadIdx.x], red[threadIdx.x + o]);
    __syncthreads();
  }
  mx = red[0]; __syncthreads();
  float s = 0.f;
#pragma unroll
  for (int i = 0; i < 8; i++) { v[i] = expf(v[i] - mx); s += v[i]; }
  red[threadIdx.x] = s; __syncthreads();
  for (int o = 128; o > 0; o >>= 1) {
    if (threadIdx.x < o) red[threadIdx.x] += red[threadIdx.x + o];
    __syncthreads();
  }
  float inv = 1.0f / red[0];
  f16* ar = A + row * 2048;
#pragma unroll
  for (int i = 0; i < 8; i++) ar[threadIdx.x + i * 256] = (f16)(v[i] * inv);
}

__global__ __launch_bounds__(256) void softmax_geo(
    const float* __restrict__ S, const float* __restrict__ qn,
    const float* __restrict__ kn, const f16* __restrict__ dbias,
    const float* __restrict__ dist, f16* __restrict__ A) {
  __shared__ float red[256];
  int r = blockIdx.x;               // [0, 32*1024)
  int z = r >> 10, m = r & 1023;
  int b = z >> 4, h = z & 15;
  const float* sr = S + (size_t)r * 1024;
  const f16* dbr = dbias + ((size_t)(b * 16 + h) * 1024 + m) * 1024;
  const float* dr = dist + ((size_t)b * 1024 + m) * 1024;
  float qterm = -0.5f * qn[(b * 1024 + m) * 16 + h];
  float v[4]; float mx = -3.4e38f;
#pragma unroll
  for (int i = 0; i < 4; i++) {
    int n = threadIdx.x + i * 256;
    float t = sr[n] + qterm - 0.5f * kn[(b * 1024 + n) * 16 + h] + (float)dbr[n];
    t = (dr[n] < 15.0f) ? t : -1e9f;
    v[i] = t; mx = fmaxf(mx, t);
  }
  red[threadIdx.x] = mx; __syncthreads();
  for (int o = 128; o > 0; o >>= 1) {
    if (threadIdx.x < o) red[threadIdx.x] = fmaxf(red[threadIdx.x], red[threadIdx.x + o]);
    __syncthreads();
  }
  mx = red[0]; __syncthreads();
  float s = 0.f;
#pragma unroll
  for (int i = 0; i < 4; i++) { v[i] = expf(v[i] - mx); s += v[i]; }
  red[threadIdx.x] = s; __syncthreads();
  for (int o = 128; o > 0; o >>= 1) {
    if (threadIdx.x < o) red[threadIdx.x] += red[threadIdx.x + o];
    __syncthreads();
  }
  float inv = 1.0f / red[0];
  f16* ar = A + (size_t)r * 1024;
#pragma unroll
  for (int i = 0; i < 4; i++) ar[threadIdx.x + i * 256] = (f16)(v[i] * inv);
}

__global__ void rowsumsq(const f16* __restrict__ X, float* __restrict__ out) {
  int i = blockIdx.x * blockDim.x + threadIdx.x;
  if (i >= 2048 * 16) return;
  int node = i >> 4, h = i & 15;
  const f16* r = X + (size_t)node * 1024 + h * 64;
  float s = 0.f;
  for (int k = 0; k < 64; k++) { float v = (float)r[k]; s += v * v; }
  out[i] = s;
}

__global__ __launch_bounds__(256) void dist_dbias(
    const float* __restrict__ coords_u, const float* __restrict__ dw1,
    const float* __restrict__ db1, const float* __restrict__ dw2,
    const float* __restrict__ db2, float* __restrict__ dist,
    f16* __restrict__ dbias) {
  __shared__ float sw1[32], sb1[32], sw2[512], sb2[16];
  if (threadIdx.x < 32) { sw1[threadIdx.x] = dw1[threadIdx.x]; sb1[threadIdx.x] = db1[threadIdx.x]; }
  if (threadIdx.x < 16) sb2[threadIdx.x] = db2[threadIdx.x];
  for (int i = threadIdx.x; i < 512; i += 256) sw2[i] = dw2[i];
  __syncthreads();
  size_t p = (size_t)blockIdx.x * 256 + threadIdx.x;   // < 2*1024*1024
  int b = (int)(p >> 20); int rem = (int)(p & ((1 << 20) - 1));
  int i = rem >> 10, j = rem & 1023;
  int ni = b * 1024 + i, nj = b * 1024 + j;
  float dx = coords_u[ni * 3 + 0] - coords_u[nj * 3 + 0];
  float dy = coords_u[ni * 3 + 1] - coords_u[nj * 3 + 1];
  float dz = coords_u[ni * 3 + 2] - coords_u[nj * 3 + 2];
  float d = sqrtf(dx * dx + dy * dy + dz * dz);
  dist[p] = d;
  float outh[16];
#pragma unroll
  for (int h = 0; h < 16; h++) outh[h] = sb2[h];
  for (int k = 0; k < 32; k++) {
    float hv = fmaxf(d * sw1[k] + sb1[k], 0.f);
#pragma unroll
    for (int h = 0; h < 16; h++) outh[h] += hv * sw2[k * 16 + h];
  }
#pragma unroll
  for (int h = 0; h < 16; h++)
    dbias[((size_t)(b * 16 + h) * 1024 + i) * 1024 + j] = (f16)outh[h];
}

__global__ void coords_update(const f16* __restrict__ hcm, const float* __restrict__ w2,
                              const float* __restrict__ b2, const float* __restrict__ temp,
                              const float* __restrict__ coords, float* __restrict__ coords_u) {
  int i = blockIdx.x * blockDim.x + threadIdx.x;
  if (i >= 2048) return;
  float a0 = b2[0], a1 = b2[1], a2 = b2[2];
  const f16* hr = hcm + (size_t)i * 256;
  for (int k = 0; k < 256; k++) {
    float hv = (float)hr[k];
    a0 += hv * w2[k * 3 + 0]; a1 += hv * w2[k * 3 + 1]; a2 += hv * w2[k * 3 + 2];
  }
  float s = 0.1f / (1.0f + expf(-temp[0]));
  coords_u[i * 3 + 0] = coords[i * 3 + 0] + tanhf(a0) * s;
  coords_u[i * 3 + 1] = coords[i * 3 + 1] + tanhf(a1) * s;
  coords_u[i * 3 + 2] = coords[i * 3 + 2] + tanhf(a2) * s;
}

// ---------------------------------------------------------------------------
// Host side
// ---------------------------------------------------------------------------
static inline void launch_gemm(hipStream_t st, const GemmP& p, int nz) {
  dim3 b(256);
  if (p.N >= 64) {
    dim3 g((p.M + 127) / 128, (p.N + 63) / 64, nz);
    gemm16<4><<<g, b, 0, st>>>(p);
  } else if (p.N == 48) {
    dim3 g((p.M + 127) / 128, 1, nz);
    gemm16<3><<<g, b, 0, st>>>(p);
  } else {
    dim3 g((p.M + 127) / 128, 1, nz);
    gemm16<2><<<g, b, 0, st>>>(p);
  }
}

extern "C" void kernel_launch(void* const* d_in, const int* in_sizes, int n_in,
                              void* d_out, int out_size, void* d_ws, size_t ws_size,
                              hipStream_t stream) {
  (void)in_sizes; (void)n_in; (void)out_size; (void)ws_size;
  const float* x        = (const float*)d_in[0];
  const float* coords   = (const float*)d_in[1];
  const int*   eidx     = (const int*)d_in[2];
  const int*   etyp     = (const int*)d_in[3];
  const float* edge_emb = (const float*)d_in[4];
  const float* msg_w1   = (const float*)d_in[5];
  const float* msg_b1   = (const float*)d_in[6];
  const float* msg_ln_g = (const float*)d_in[7];
  const float* msg_ln_b = (const float*)d_in[8];
  const float* msg_w2   = (const float*)d_in[9];
  const float* msg_b2   = (const float*)d_in[10];
  const float* node_w1  = (const float*)d_in[11];
  const float* node_b1  = (const float*)d_in[12];
  const float* node_ln_g= (const float*)d_in[13];
  const float* node_ln_b= (const float*)d_in[14];
  const float* node_w2  = (const float*)d_in[15];
  const float* node_b2  = (const float*)d_in[16];
  const float* ca_wq    = (const float*)d_in[17];
  const float* ca_wk    = (const float*)d_in[18];
  const float* ca_wv    = (const float*)d_in[19];
  const float* ca_wo    = (const float*)d_in[20];
  const float* ca_bq    = (const float*)d_in[21];
  const float* ca_bk    = (const float*)d_in[22];
  const float* ca_bv    = (const float*)d_in[23];
  const float* ca_bo    = (const float*)d_in[24];
  const float* cm_w1    = (const float*)d_in[25];
  const float* cm_b1    = (const float*)d_in[26];
  const float* cm_w2    = (const float*)d_in[27];
  const float* cm_b2    = (const float*)d_in[28];
  const float* temp     = (const float*)d_in[29];
  const float* g_n1g    = (const float*)d_in[30];
  const float* g_n1b    = (const float*)d_in[31];
  const float* g_n2g    = (const float*)d_in[32];
  const float* g_n2b    = (const float*)d_in[33];
  const float* g_wq     = (const float*)d_in[34];
  const float* g_wk     = (const float*)d_in[35];
  const float* g_wv     = (const float*)d_in[36];
  const float* g_wqp    = (const float*)d_in[37];
  const float* g_bqp    = (const float*)d_in[38];
  const float* g_wkp    = (const float*)d_in[39];
  const float* g_bkp    = (const float*)d_in[40];
  const float* g_wvp    = (const float*)d_in[41];
  const float* g_bvp    = (const float*)d_in[42];
  const float* g_dw1    = (const float*)d_in[43];
  const float* g_db1    = (const float*)d_in[44];
  const float* g_dw2    = (const float*)d_in[45];
  const float* g_db2    = (const float*)d_in[46];
  const float* g_wo     = (const float*)d_in[47];
  const float* g_bo     = (const float*)d_in[48];

  // ---- workspace bump allocator ----
  char* base = (char*)d_ws; size_t off = 0;
  auto alloc = [&](size_t bytes) -> void* {
    void* p = base + off; off = (off + bytes + 255) & ~(size_t)255; return p;
  };
  // f16 transposed weights [N, Kpad]
  f16* msg_w1t = (f16*)alloc(1024ull * 1088 * 2);
  f16* msg_w2t = (f16*)alloc(512ull * 1024 * 2);
  f16* node_w1t= (f16*)alloc(1024ull * 1024 * 2);
  f16* node_w2t= (f16*)alloc(512ull * 1024 * 2);
  f16* ca_wqt  = (f16*)alloc(512ull * 512 * 2);
  f16* ca_wkt  = (f16*)alloc(512ull * 512 * 2);
  f16* ca_wvt  = (f16*)alloc(512ull * 512 * 2);
  f16* ca_wot  = (f16*)alloc(512ull * 512 * 2);
  f16* cm_w1t  = (f16*)alloc(256ull * 512 * 2);
  f16* g_wqt   = (f16*)alloc(512ull * 512 * 2);
  f16* g_wkt   = (f16*)alloc(512ull * 512 * 2);
  f16* g_wvt   = (f16*)alloc(512ull * 512 * 2);
  f16* g_wqpt  = (f16*)alloc(768ull * 512 * 2);
  f16* g_wkpt  = (f16*)alloc(768ull * 512 * 2);
  f16* g_wvpt  = (f16*)alloc(768ull * 512 * 2);
  f16* g_wot   = (f16*)alloc(512ull * 1280 * 2);
  // big reusable regions
  char* regA = (char*)alloc(142606336ull);   // mi16 [E,1088]f16  -> scoresG [32,1024,1024]f32
  char* regB = (char*)alloc(134217728ull);   // h16 [E,1024]f16 -> aw_ca -> awG(67MB)+dbias(67MB)
  char* regC = (char*)alloc(67108864ull);    // scores_ca [4,2048,2048]f32 -> dist [2,1024,1024]f32
  f16*   mi16      = (f16*)regA;
  float* scoresG   = (float*)regA;
  f16*   h16       = (f16*)regB;
  f16*   aw_ca     = (f16*)regB;
  f16*   awG       = (f16*)regB;
  f16*   dbias16   = (f16*)(regB + 67108864ull);
  float* scores_ca = (float*)regC;
  float* distb     = (float*)regC;
  // small buffers
  float* agg     = (float*)alloc(2048ull * 512 * 4);
  f16*   xcat16  = (f16*)alloc(2048ull * 1024 * 2);
  f16*   hn16    = (f16*)alloc(2048ull * 1024 * 2);
  float* xu      = (float*)alloc(2048ull * 512 * 4);
  f16*   xu16    = (f16*)alloc(2048ull * 512 * 2);
  f16*   q16     = (f16*)alloc(2048ull * 512 * 2);
  f16*   k16     = (f16*)alloc(2048ull * 512 * 2);
  f16*   vT16    = (f16*)alloc(512ull * 2048 * 2);
  f16*   ctx16   = (f16*)alloc(2048ull * 512 * 2);
  f16*   ctxo16  = (f16*)alloc(2048ull * 512 * 2);
  f16*   hcm16   = (f16*)alloc(2048ull * 256 * 2);
  float* coords_u= (float*)alloc(2048ull * 3 * 4);
  f16*   xn16    = (f16*)alloc(2048ull * 512 * 2);
  f16*   qp16    = (f16*)alloc(2048ull * 1024 * 2);
  f16*   kp16    = (f16*)alloc(2048ull * 1024 * 2);
  f16*   vpT16   = (f16*)alloc(768ull * 2048 * 2);
  float* qn      = (float*)alloc(2048ull * 16 * 4);
  float* kn      = (float*)alloc(2048ull * 16 * 4);
  f16*   cat16   = (f16*)alloc(2048ull * 1280 * 2);
  float* outpre  = (float*)alloc(2048ull * 512 * 4);

  auto cvtw = [&](const float* W, f16* Wt, int K, int Nc, int Kpad) {
    int tot = Nc * Kpad;
    cvt_w<<<(tot + 255) / 256, 256, 0, stream>>>(W, Wt, K, Nc, Kpad);
  };
  // ---- phase 0: weights -> f16 transposed ----
  cvtw(msg_w1, msg_w1t, 1057, 1024, 1088);
  cvtw(msg_w2, msg_w2t, 1024, 512, 1024);
  cvtw(node_w1, node_w1t, 1024, 1024, 1024);
  cvtw(node_w2, node_w2t, 1024, 512, 1024);
  cvtw(ca_wq, ca_wqt, 512, 512, 512);
  cvtw(ca_wk, ca_wkt, 512, 512, 512);
  cvtw(ca_wv, ca_wvt, 512, 512, 512);
  cvtw(ca_wo, ca_wot, 512, 512, 512);
  cvtw(cm_w1, cm_w1t, 512, 256, 512);
  cvtw(g_wq, g_wqt, 512, 512, 512);
  cvtw(g_wk, g_wkt, 512, 512, 512);
  cvtw(g_wv, g_wvt, 512, 512, 512);
  cvtw(g_wqp, g_wqpt, 512, 768, 512);
  cvtw(g_wkp, g_wkpt, 512, 768, 512);
  cvtw(g_wvp, g_wvpt, 512, 768, 512);
  cvtw(g_wo, g_wot, 1280, 512, 1280);

  // ---- phase 1: edge features + edge MLP (dominant GEMMs) ----
  build_mi<<<65536, 256, 0, stream>>>(x, coords, eidx, etyp, edge_emb, mi16);
  gemm_rowln<<<65536 / 16, 256, 0, stream>>>(mi16, msg_w1t, msg_b1, msg_ln_g,
                                             msg_ln_b, h16, 1088, 1088);
  hipMemsetAsync(agg, 0, 2048ull * 512 * 4, stream);
  {
    GemmP p{}; p.A = h16; p.B = msg_w2t; p.Cf = agg; p.bias = msg_b2;
    p.sidx = eidx + 65536;
    p.M = 65536; p.N = 512; p.K = 1024; p.lda = 1024; p.ldb = 1024; p.ldc = 512;
    p.zinner = 1; p.scale = 1.f; p.flags = F_SCATTER;
    launch_gemm(stream, p, 1);
  }
  // ---- phase 2: node MLP ----
  build_xcat<<<(2048 * 1024 + 255) / 256, 256, 0, stream>>>(x, agg, xcat16);
  gemm_rowln<<<2048 / 16, 256, 0, stream>>>(xcat16, node_w1t, node_b1, node_ln_g,
                                            node_ln_b, hn16, 1024, 1024);
  {
    GemmP p{}; p.A = hn16; p.B = node_w2t; p.Cf = xu; p.bias = node_b2; p.resid = x;
    p.M = 2048; p.N = 512; p.K = 1024; p.lda = 1024; p.ldb = 1024; p.ldc = 512;
    p.zinner = 1; p.scale = 1.f; p.flags = F_RESID;
    launch_gemm(stream, p, 1);
  }
  cvt_f32_f16<<<(2048 * 512 + 255) / 256, 256, 0, stream>>>(xu, xu16, 2048 * 512);

  // ---- phase 3: coord attention (4 heads, hd=128, seq=2048) ----
  {
    GemmP p{}; p.A = xu16; p.B = ca_wqt; p.Ch = q16; p.bias = ca_bq;
    p.M = 2048; p.N = 512; p.K = 512; p.lda = 512; p.ldb = 512; p.ldc = 512;
    p.zinner = 1; p.scale = 1.f; p.flags = F_OUTF16;
    launch_gemm(stream, p, 1);
    p.B = ca_wkt; p.Ch = k16; p.bias = ca_bk; launch_gemm(stream, p, 1);
    p.B = ca_wvt; p.Ch = vT16; p.bias = ca_bv; p.ldc = 2048;
    p.flags = F_OUTF16 | F_TRANS; launch_gemm(stream, p, 1);
  }
  {
    GemmP p{}; p.A = q16; p.B = k16; p.Cf = scores_ca;
    p.M = 2048; p.N = 2048; p.K = 128; p.lda = 512; p.ldb = 512; p.ldc = 2048;
    p.sAin = 128; p.sBin = 128; p.sCin = 2048ll * 2048;
    p.zinner = 4; p.scale = 0.08838834764831845f; p.flags = 0;
    launch_gemm(stream, p, 4);
  }
  softmax_ca<<<4 * 2048, 256, 0, stream>>>(scores_ca, aw_ca);
  {
    GemmP p{}; p.A = aw_ca; p.B = vT16; p.Ch = ctx16;
    p.M = 2048; p.N = 128; p.K = 2048; p.lda = 2048; p.ldb = 2048; p.ldc = 512;
    p.sAin = 2048ll * 2048; p.sBin = 128ll * 2048; p.sCin = 128;
    p.zinner = 4; p.scale = 1.f; p.flags = F_OUTF16;
    launch_gemm(stream, p, 4);
  }
  {
    GemmP p{}; p.A = ctx16; p.B = ca_wot; p.Ch = ctxo16; p.bias = ca_bo;
    p.M = 2048; p.N = 512; p.K = 512; p.lda = 512; p.ldb = 512; p.ldc = 512;
    p.zinner = 1; p.scale = 1.f; p.flags = F_OUTF16;
    launch_gemm(stream, p, 1);
  }
  {
    GemmP p{}; p.A = ctxo16; p.B = cm_w1t; p.Ch = hcm16; p.bias = cm_b1;
    p.M = 2048; p.N = 256; p.K = 512; p.lda = 512; p.ldb = 512; p.ldc = 256;
    p.zinner = 1; p.scale = 1.f; p.flags = F_OUTF16 | F_GELU;
    launch_gemm(stream, p, 1);
  }
  coords_update<<<8, 256, 0, stream>>>(hcm16, cm_w2, cm_b2, temp, coords, coords_u);

  // ---- phase 4: geometric self-attention ----
  ln_rows<<<2048, 256, 0, stream>>>(xu, g_n1g, g_n1b, nullptr, xn16, 512);
  {
    GemmP p{}; p.A = xn16; p.B = g_wqt; p.Ch = q16;
    p.M = 2048; p.N = 512; p.K = 512; p.lda = 512; p.ldb = 512; p.ldc = 512;
    p.zinner = 1; p.scale = 1.f; p.flags = F_OUTF16;
    launch_gemm(stream, p, 1);
    p.B = g_wkt; p.Ch = k16; launch_gemm(stream, p, 1);
    p.B = g_wvt; p.Ch = vT16; p.ldc = 2048; p.flags = F_OUTF16 | F_TRANS;
    launch_gemm(stream, p, 1);
  }
  hipMemsetAsync(qp16, 0, 2048ull * 1024 * 2, stream);
  hipMemsetAsync(kp16, 0, 2048ull * 1024 * 2, stream);
  {
    GemmP p{}; p.A = xn16; p.B = g_wqpt; p.Ch = qp16; p.bias = g_bqp;
    p.coord = coords_u;
    p.M = 2048; p.N = 768; p.K = 512; p.lda = 512; p.ldb = 512; p.ldc = 1024;
    p.zinner = 1; p.scale = 1.f; p.flags = F_OUTF16 | F_QPAD;
    launch_gemm(stream, p, 1);
    p.B = g_wkpt; p.Ch = kp16; p.bias = g_bkp; launch_gemm(stream, p, 1);
    p.B = g_wvpt; p.Ch = vpT16; p.bias = g_bvp; p.coord = nullptr;
    p.ldc = 2048; p.flags = F_OUTF16 | F_TRANS; launch_gemm(stream, p, 1);
  }
  rowsumsq<<<(2048 * 16 + 255) / 256, 256, 0, stream>>>(qp16, qn);
  rowsumsq<<<(2048 * 16 + 255) / 256, 256, 0, stream>>>(kp16, kn);
  dist_dbias<<<8192, 256, 0, stream>>>(coords_u, g_dw1, g_db1, g_dw2, g_db2,
                                       distb, dbias16);
  {  // feature attention scores, per (b,h): [1024,32] @ [1024,32]^T
    GemmP p{}; p.A = q16; p.B = k16; p.Cf = scoresG;
    p.M = 1024; p.N = 1024; p.K = 32; p.lda = 512; p.ldb = 512; p.ldc = 1024;
    p.sAin = 32; p.sAout = 1024ll * 512; p.sBin = 32; p.sBout = 1024ll * 512;
    p.sCin = 1024ll * 1024; p.sCout = 16ll * 1024 * 1024;
    p.zinner = 16; p.scale = 0.17677669529663687f; p.flags = 0;
    launch_gemm(stream, p, 32);
  }
  {  // point-score qp.kp term accumulated onto scores (K padded 48->64 with zeros)
    GemmP p{}; p.A = qp16; p.B = kp16; p.Cf = scoresG;
    p.M = 1024; p.N = 1024; p.K = 64; p.lda = 1024; p.ldb = 1024; p.ldc = 1024;
    p.sAin = 64; p.sAout = 1024ll * 1024; p.sBin = 64; p.sBout = 1024ll * 1024;
    p.sCin = 1024ll * 1024; p.sCout = 16ll * 1024 * 1024;
    p.zinner = 16; p.scale = 1.f; p.flags = F_ACCUM;
    launch_gemm(stream, p, 32);
  }
  softmax_geo<<<32 * 1024, 256, 0, stream>>>(scoresG, qn, kn, dbias16, distb, awG);
  {  // af = aw @ v  -> cat16[:, 0:512]
    GemmP p{}; p.A = awG; p.B = vT16; p.Ch = cat16;
    p.M = 1024; p.N = 32; p.K = 1024; p.lda = 1024; p.ldb = 2048; p.ldc = 1280;
    p.sAin = 1024ll * 1024; p.sAout = 16ll * 1024 * 1024;
    p.sBin = 32ll * 2048; p.sBout = 1024;
    p.sCin = 32; p.sCout = 1024ll * 1280;
    p.zinner = 16; p.scale = 1.f; p.flags = F_OUTF16;
    launch_gemm(stream, p, 32);
  }
  {  // ap = aw @ vp -> cat16[:, 512:1280]
    GemmP p{}; p.A = awG; p.B = vpT16; p.Ch = cat16 + 512;
    p.M = 1024; p.N = 48; p.K = 1024; p.lda = 1024; p.ldb = 2048; p.ldc = 1280;
    p.sAin = 1024ll * 1024; p.sAout = 16ll * 1024 * 1024;
    p.sBin = 48ll * 2048; p.sBout = 1024;
    p.sCin = 48; p.sCout = 1024ll * 1280;
    p.zinner = 16; p.scale = 1.f; p.flags = F_OUTF16;
    launch_gemm(stream, p, 32);
  }
  {  // output projection + residual
    GemmP p{}; p.A = cat16; p.B = g_wot; p.Cf = outpre; p.bias = g_bo; p.resid = xu;
    p.M = 2048; p.N = 512; p.K = 1280; p.lda = 1280; p.ldb = 1280; p.ldc = 512;
    p.zinner = 1; p.scale = 1.f; p.flags = F_RESID;
    launch_gemm(stream, p, 1);
  }
  ln_rows<<<2048, 256, 0, stream>>>(outpre, g_n2g, g_n2b, (float*)d_out, nullptr, 512);
}
